// Qwen3Attention_83708912599569
// MI455X (gfx1250) — compile-verified, matrix operand order checked by reading, not run on hardware
//
#include <hip/hip_runtime.h>
#include <hip/hip_bf16.h>

// ---------- problem constants ----------
constexpr int kS   = 2048;
constexpr int kHID = 4096;
constexpr int kNH  = 32;
constexpr int kNKV = 8;
constexpr int kHD  = 128;
constexpr float kScale = 0.08838834764831845f;   // 1/sqrt(128)
constexpr float kEps   = 1e-6f;

// ---------- vector types ----------
typedef __attribute__((ext_vector_type(16))) __bf16 bf16x16;
typedef __attribute__((ext_vector_type(8)))  __bf16 bf16x8;
typedef __attribute__((ext_vector_type(8)))  float  f32x8;

// ---------- WMMA wrapper ----------
__device__ __forceinline__ f32x8 wmma_bf16(bf16x16 a, bf16x16 b, f32x8 c) {
    // (neg_a, A, neg_b, B, c_mod, C, reuse_a, reuse_b)
    return __builtin_amdgcn_wmma_f32_16x16x32_bf16(false, a, false, b, (short)0, c, false, false);
}

// Load a 16x32 A-fragment (or 32x16 B-fragment) row for this lane.
// `base` points at element K=0 of this lane's row/column (row-contiguous),
// half = lane/16. Per ISA layout: chunks at K = half*8..+7 and K = 16+half*8..+7.
__device__ __forceinline__ bf16x16 load_ab(const __bf16* base, int half) {
    bf16x8 c0 = *(const bf16x8*)(base + half * 8);
    bf16x8 c1 = *(const bf16x8*)(base + 16 + half * 8);
    bf16x16 r;
#pragma unroll
    for (int i = 0; i < 8; ++i) { r[i] = c0[i]; r[i + 8] = c1[i]; }
    return r;
}

// ---------- wave reductions (wave32; 16-lane half-groups use xor masks 1..8) ----------
__device__ __forceinline__ float half_max16(float v) {
#pragma unroll
    for (int m = 1; m < 16; m <<= 1) v = fmaxf(v, __shfl_xor(v, m, 32));
    return v;
}
__device__ __forceinline__ float half_sum16(float v) {
#pragma unroll
    for (int m = 1; m < 16; m <<= 1) v += __shfl_xor(v, m, 32);
    return v;
}
__device__ __forceinline__ float wave_sum32(float v) {
#pragma unroll
    for (int m = 1; m < 32; m <<= 1) v += __shfl_xor(v, m, 32);
    return v;
}

// ---------- fp32 -> bf16 cast ----------
__global__ void cast_f32_bf16(const float* __restrict__ in, __bf16* __restrict__ out, int n) {
    int i = blockIdx.x * blockDim.x + threadIdx.x;
    if (i < n) out[i] = (__bf16)in[i];
}

// ---------- V: [S][NKV*HD] fp32 -> [NKV][HD][S] bf16 ----------
__global__ void v_transpose_cast(const float* __restrict__ V, __bf16* __restrict__ Vt) {
    int idx = blockIdx.x * blockDim.x + threadIdx.x;
    if (idx >= kS * kNKV * kHD) return;
    int d = idx % kHD;
    int h = (idx / kHD) % kNKV;
    int s = idx / (kHD * kNKV);
    Vt[((size_t)h * kHD + d) * kS + s] = (__bf16)V[idx];
}

// ---------- fused RMSNorm + RoPE:  X [S][H*HD] fp32 -> Out [H][S][HD] bf16 ----------
__global__ void __launch_bounds__(128) rmsnorm_rope(const float* __restrict__ X,
                                                    const float* __restrict__ w,
                                                    const float* __restrict__ cs,
                                                    const float* __restrict__ sn,
                                                    __bf16* __restrict__ Out, int H) {
    int wid  = (blockIdx.x * blockDim.x + threadIdx.x) >> 5;
    int lane = threadIdx.x & 31;
    int s = wid % kS;
    int h = wid / kS;
    if (h >= H) return;
    const float* x = X + (size_t)s * (H * kHD) + (size_t)h * kHD;

    float xl[4], sum = 0.f;
#pragma unroll
    for (int j = 0; j < 4; ++j) { xl[j] = x[lane + 32 * j]; sum += xl[j] * xl[j]; }
    sum = wave_sum32(sum) * (1.0f / kHD);
    float inv = rsqrtf(sum + kEps);

    float xn[4];
#pragma unroll
    for (int j = 0; j < 4; ++j) xn[j] = xl[j] * inv * w[lane + 32 * j];

    __bf16* orow = Out + ((size_t)h * kS + s) * kHD;
#pragma unroll
    for (int j = 0; j < 4; ++j) {
        int d = lane + 32 * j;
        float rot = (j < 2) ? -xn[j + 2] : xn[j - 2];   // rotate_half
        float o = xn[j] * cs[(size_t)s * kHD + d] + rot * sn[(size_t)s * kHD + d];
        orow[d] = (__bf16)o;
    }
}

// ---------- GEMM: C[M][N] = A[M][K] * B[N][K]^T  (bf16 in, fp32 out) ----------
// block = 128 threads (4 waves); wave tile 32(M) x 64(N); block tile 32 x 256.
__global__ void __launch_bounds__(128) gemm_bf16_nt(const __bf16* __restrict__ A,
                                                    const __bf16* __restrict__ B,
                                                    float* __restrict__ C,
                                                    int M, int N, int K) {
    int widx = threadIdx.x >> 5;
    int lane = threadIdx.x & 31;
    int half = lane >> 4, lr = lane & 15;
    int m0 = blockIdx.y * 32;
    int n0 = blockIdx.x * 256 + widx * 64;

    f32x8 z = {};
    f32x8 acc[2][4];
#pragma unroll
    for (int mi = 0; mi < 2; ++mi)
#pragma unroll
        for (int ni = 0; ni < 4; ++ni) acc[mi][ni] = z;

    const __bf16* arow0 = A + (size_t)(m0 + lr) * K;
    const __bf16* arow1 = A + (size_t)(m0 + 16 + lr) * K;
    const __bf16* brow[4];
#pragma unroll
    for (int ni = 0; ni < 4; ++ni) brow[ni] = B + (size_t)(n0 + ni * 16 + lr) * K;

#pragma unroll 2
    for (int k0 = 0; k0 < K; k0 += 32) {
        // stream ~512B ahead along K (rows are K-contiguous)
        __builtin_prefetch(arow0 + k0 + 256, 0, 1);
        __builtin_prefetch(arow1 + k0 + 256, 0, 1);
        bf16x16 a0 = load_ab(arow0 + k0, half);
        bf16x16 a1 = load_ab(arow1 + k0, half);
        bf16x16 b[4];
#pragma unroll
        for (int ni = 0; ni < 4; ++ni) {
            __builtin_prefetch(brow[ni] + k0 + 256, 0, 1);
            b[ni] = load_ab(brow[ni] + k0, half);
        }
#pragma unroll
        for (int ni = 0; ni < 4; ++ni) {
            acc[0][ni] = wmma_bf16(a0, b[ni], acc[0][ni]);
            acc[1][ni] = wmma_bf16(a1, b[ni], acc[1][ni]);
        }
    }

#pragma unroll
    for (int mi = 0; mi < 2; ++mi)
#pragma unroll
        for (int ni = 0; ni < 4; ++ni)
#pragma unroll
            for (int r = 0; r < 8; ++r) {
                int row = m0 + mi * 16 + r + 8 * half;
                int col = n0 + ni * 16 + lr;
                C[(size_t)row * N + col] = acc[mi][ni][r];
            }
}

// ---------- causal flash attention ----------
// Qn [NH][S][HD] bf16, Kn [NKV][S][HD] bf16, Vt [NKV][HD][S] bf16 -> O [S][NH*HD] bf16
// one wave per (head, 16-query block); key blocks of 32.
__global__ void __launch_bounds__(128) flash_attn(const __bf16* __restrict__ Qn,
                                                  const __bf16* __restrict__ Kn,
                                                  const __bf16* __restrict__ Vt,
                                                  __bf16* __restrict__ O) {
    __shared__ __bf16 lds[4 * 16 * 32];
    int widx = threadIdx.x >> 5;
    int lane = threadIdx.x & 31;
    int wid  = blockIdx.x * 4 + widx;
    int qb   = wid & (kS / 16 - 1);     // 0..127
    int h    = wid >> 7;                // 0..31
    int kvh  = h >> 2;                  // GQA group
    int half = lane >> 4, lr = lane & 15;
    int q0   = qb * 16;
    __bf16* P = lds + widx * 16 * 32;

    // Q fragments for all of HD=128 (4 K-steps of 32)
    const __bf16* qbase = Qn + ((size_t)h * kS + q0 + lr) * kHD;
    bf16x16 qa[4];
#pragma unroll
    for (int kk = 0; kk < 4; ++kk) qa[kk] = load_ab(qbase + kk * 32, half);

    f32x8 z = {};
    f32x8 acc[8];
#pragma unroll
    for (int di = 0; di < 8; ++di) acc[di] = z;
    float rmax[8], rsum[8];
#pragma unroll
    for (int r = 0; r < 8; ++r) { rmax[r] = -3e38f; rsum[r] = 0.f; }

    int nb = (qb + 2) >> 1;   // ceil((qb+1)*16 / 32)
    for (int jb = 0; jb < nb; ++jb) {
        int j0 = jb * 32;
        const __bf16* kbase = Kn + ((size_t)kvh * kS + j0 + lr) * kHD;
        // prefetch next key block (rows contiguous along S)
        __builtin_prefetch(kbase + (size_t)32 * kHD, 0, 1);

        f32x8 s0 = z, s1 = z;
#pragma unroll
        for (int kk = 0; kk < 4; ++kk) {
            bf16x16 k0f = load_ab(kbase + kk * 32, half);
            bf16x16 k1f = load_ab(kbase + (size_t)16 * kHD + kk * 32, half);
            s0 = wmma_bf16(qa[kk], k0f, s0);
            s1 = wmma_bf16(qa[kk], k1f, s1);
        }

        // scale + causal mask + online softmax
#pragma unroll
        for (int r = 0; r < 8; ++r) {
            int qrow = q0 + r + 8 * half;
            float sc0 = (j0 + lr      <= qrow) ? s0[r] * kScale : -3e38f;
            float sc1 = (j0 + 16 + lr <= qrow) ? s1[r] * kScale : -3e38f;
            float mx = half_max16(fmaxf(sc0, sc1));
            float nm = fmaxf(rmax[r], mx);
            float corr = __expf(rmax[r] - nm);
            rmax[r] = nm;
            float p0 = __expf(sc0 - nm);
            float p1 = __expf(sc1 - nm);
            rsum[r] = rsum[r] * corr + half_sum16(p0 + p1);
#pragma unroll
            for (int di = 0; di < 8; ++di) acc[di][r] *= corr;
            P[(r + 8 * half) * 32 + lr]      = (__bf16)p0;
            P[(r + 8 * half) * 32 + 16 + lr] = (__bf16)p1;
        }
        asm volatile("s_wait_dscnt 0" ::: "memory");  // P stores visible before re-read

        // reload P in A-layout (row = lr, K = key index within block)
        bf16x16 pa = load_ab(P + lr * 32, half);
#pragma unroll
        for (int di = 0; di < 8; ++di) {
            const __bf16* vb = Vt + ((size_t)kvh * kHD + di * 16 + lr) * kS + j0;
            bf16x16 vf = load_ab(vb, half);
            acc[di] = wmma_bf16(pa, vf, acc[di]);
        }
    }

    // normalize + store
#pragma unroll
    for (int r = 0; r < 8; ++r) {
        float inv = 1.0f / rsum[r];
        int row = q0 + r + 8 * half;
        __bf16* orow = O + (size_t)row * (kNH * kHD) + (size_t)h * kHD;
#pragma unroll
        for (int di = 0; di < 8; ++di)
            orow[di * 16 + lr] = (__bf16)(acc[di][r] * inv);
    }
}

// ---------- host-side launcher ----------
extern "C" void kernel_launch(void* const* d_in, const int* in_sizes, int n_in,
                              void* d_out, int out_size, void* d_ws, size_t ws_size,
                              hipStream_t stream) {
    (void)in_sizes; (void)n_in; (void)out_size; (void)ws_size;
    const float* hidden = (const float*)d_in[0];
    const float* cosd   = (const float*)d_in[1];
    const float* sind   = (const float*)d_in[2];
    const float* Wq     = (const float*)d_in[3];
    const float* Wk     = (const float*)d_in[4];
    const float* Wv     = (const float*)d_in[5];
    const float* Wo     = (const float*)d_in[6];
    const float* qnw    = (const float*)d_in[7];
    const float* knw    = (const float*)d_in[8];
    float* out = (float*)d_out;

    // workspace carve-up
    char* ws = (char*)d_ws;
    auto alloc = [&](size_t bytes) { char* p = ws; ws += (bytes + 255) & ~size_t(255); return p; };
    __bf16* hb  = (__bf16*)alloc((size_t)kS * kHID * 2);          // hidden bf16
    __bf16* wqb = (__bf16*)alloc((size_t)kNH * kHD * kHID * 2);   // Wq bf16
    __bf16* wkb = (__bf16*)alloc((size_t)kNKV * kHD * kHID * 2);  // Wk bf16
    __bf16* wvb = (__bf16*)alloc((size_t)kNKV * kHD * kHID * 2);  // Wv bf16
    __bf16* wob = (__bf16*)alloc((size_t)kHID * kNH * kHD * 2);   // Wo bf16
    float*  Qf  = (float*) alloc((size_t)kS * kNH * kHD * 4);     // Q fp32
    float*  Kf  = (float*) alloc((size_t)kS * kNKV * kHD * 4);    // K fp32
    float*  Vf  = (float*) alloc((size_t)kS * kNKV * kHD * 4);    // V fp32
    __bf16* Qn  = (__bf16*)alloc((size_t)kNH * kS * kHD * 2);     // normed+rope Q
    __bf16* Kn  = (__bf16*)alloc((size_t)kNKV * kS * kHD * 2);    // normed+rope K
    __bf16* Vt  = (__bf16*)alloc((size_t)kNKV * kHD * kS * 2);    // V transposed
    __bf16* Ob  = (__bf16*)alloc((size_t)kS * kNH * kHD * 2);     // attention out

    auto cast = [&](const float* src, __bf16* dst, int n) {
        cast_f32_bf16<<<(n + 255) / 256, 256, 0, stream>>>(src, dst, n);
    };
    cast(hidden, hb, kS * kHID);
    cast(Wq, wqb, kNH * kHD * kHID);
    cast(Wk, wkb, kNKV * kHD * kHID);
    cast(Wv, wvb, kNKV * kHD * kHID);
    cast(Wo, wob, kHID * kNH * kHD);

    // QKV projections: C = hidden * W^T
    gemm_bf16_nt<<<dim3((kNH * kHD) / 256, kS / 32), 128, 0, stream>>>(hb, wqb, Qf, kS, kNH * kHD, kHID);
    gemm_bf16_nt<<<dim3((kNKV * kHD) / 256, kS / 32), 128, 0, stream>>>(hb, wkb, Kf, kS, kNKV * kHD, kHID);
    gemm_bf16_nt<<<dim3((kNKV * kHD) / 256, kS / 32), 128, 0, stream>>>(hb, wvb, Vf, kS, kNKV * kHD, kHID);

    // per-head RMSNorm + RoPE (fp32 -> bf16, head-major layout)
    rmsnorm_rope<<<(kS * kNH) / 4, 128, 0, stream>>>(Qf, qnw, cosd, sind, Qn, kNH);
    rmsnorm_rope<<<(kS * kNKV) / 4, 128, 0, stream>>>(Kf, knw, cosd, sind, Kn, kNKV);

    // V -> [NKV][HD][S] bf16
    v_transpose_cast<<<(kS * kNKV * kHD + 255) / 256, 256, 0, stream>>>(Vf, Vt);

    // causal flash attention
    flash_attn<<<(kNH * (kS / 16)) / 4, 128, 0, stream>>>(Qn, Kn, Vt, Ob);

    // output projection: out = O * Wo^T
    gemm_bf16_nt<<<dim3(kHID / 256, kS / 32), 128, 0, stream>>>(Ob, wob, out, kS, kHID, kNH * kHD);
}
